// TensorcontextSeq2Mat_10539849744801
// MI455X (gfx1250) — matrix operand order repaired and checked by reading, not compiled, not run on hardware
//
#include <hip/hip_runtime.h>
#include <hip/hip_bf16.h>

// Shapes (fixed by the reference): B=2, L=128, H=512, D=8
static constexpr int kB  = 2;
static constexpr int kL  = 128;
static constexpr int kH  = 512;
static constexpr int kD  = 8;
static constexpr int kWW = 3 * kH + kD;       // 1544 columns in W_w
static constexpr int kRows = kB * kL * kL;    // 32768 GEMM rows (b,m,n)

typedef __attribute__((ext_vector_type(16))) _Float16 v16h;
typedef __attribute__((ext_vector_type(8)))  _Float16 v8h;
typedef __attribute__((ext_vector_type(8)))  float    v8f;

__device__ __forceinline__ float gelu_exact(float g) {
    return 0.5f * g * (1.0f + erff(g * 0.70710678118654752f));
}

// ---------------------------------------------------------------------------
// Phase 1: ctx[b,m,n,h] = max_{k in [min(m,n),max(m,n)]} x[b,k,h], stored f16
// in GEMM A layout: ctxA[(b*L+m)*L + n][h].  One block per (b,m); thread = h.
// ---------------------------------------------------------------------------
__global__ void __launch_bounds__(512)
ctx_kernel(const float* __restrict__ x, _Float16* __restrict__ ctxA)
{
    const int b = blockIdx.x >> 7;       // / kL
    const int m = blockIdx.x & (kL - 1);
    const int h = threadIdx.x;
    const float* xb = x + (size_t)b * kL * kH;
    _Float16* base = ctxA + ((size_t)(b * kL + m) * kL) * kH + h;

    const float r0 = xb[(size_t)m * kH + h];
    base[(size_t)m * kH] = (_Float16)r0;
    float r = r0;
    for (int n = m + 1; n < kL; ++n) {                 // upward span
        r = fmaxf(r, xb[(size_t)n * kH + h]);
        base[(size_t)n * kH] = (_Float16)r;
    }
    r = r0;
    for (int n = m - 1; n >= 0; --n) {                 // downward span
        r = fmaxf(r, xb[(size_t)n * kH + h]);
        base[(size_t)n * kH] = (_Float16)r;
    }
}

// ---------------------------------------------------------------------------
// Pack W3 = W_w[:, 2H:3H] transposed to f16: Bp[n][k] = W_w[n, 2H+k]
// (column n contiguous in k -> matches WMMA B-fragment per-lane layout)
// ---------------------------------------------------------------------------
__global__ void pack_w3_kernel(const float* __restrict__ Ww, _Float16* __restrict__ Bp)
{
    int idx = blockIdx.x * blockDim.x + threadIdx.x;
    if (idx >= kH * kH) return;
    int n = idx >> 9, k = idx & (kH - 1);
    Bp[(size_t)n * kH + k] = (_Float16)Ww[(size_t)n * kWW + 2 * kH + k];
}

// ---------------------------------------------------------------------------
// Arow[b,m,h'] = x[b,m,:]·W1ᵀ + W_b[h'] ; Brow[b,n,h'] = y[b,n,:]·W2ᵀ
// 512 blocks: [0,256) -> Arow, [256,512) -> Brow. thread = h'.
// ---------------------------------------------------------------------------
__global__ void __launch_bounds__(512)
row_gemm_kernel(const float* __restrict__ x, const float* __restrict__ y,
                const float* __restrict__ Ww, const float* __restrict__ Wb,
                float* __restrict__ Arow, float* __restrict__ Brow)
{
    __shared__ float s[kH];
    const int isB = blockIdx.x >= kB * kL;
    const int idx = blockIdx.x & (kB * kL - 1);
    const int hp  = threadIdx.x;
    const float* src = (isB ? y : x) + (size_t)idx * kH;
    s[hp] = src[hp];
    __syncthreads();
    const float* wrow = Ww + (size_t)hp * kWW + (isB ? kH : 0);
    float acc = isB ? 0.0f : Wb[hp];
    for (int h = 0; h < kH; ++h) acc = fmaf(s[h], wrow[h], acc);
    (isB ? Brow : Arow)[(size_t)idx * kH + hp] = acc;
}

// ---------------------------------------------------------------------------
// xV[b,m,k,q] = sum_p x[b,m,p] * V[k,p,q].  Block per (b*L+m, k); thread = q.
// ---------------------------------------------------------------------------
__global__ void __launch_bounds__(512)
xv_kernel(const float* __restrict__ x, const float* __restrict__ V,
          float* __restrict__ xV)
{
    __shared__ float s[kH];
    const int bm = blockIdx.x >> 3;
    const int k  = blockIdx.x & (kD - 1);
    const int q  = threadIdx.x;
    s[q] = x[(size_t)bm * kH + q];
    __syncthreads();
    const float* vk = V + (size_t)k * kH * kH + q;
    float acc = 0.f;
    for (int p = 0; p < kH; ++p) acc = fmaf(s[p], vk[(size_t)p * kH], acc);
    xV[((size_t)bm * kD + k) * kH + q] = acc;
}

// ---------------------------------------------------------------------------
// xvy[b,m,n,k] = sum_q xV[b,m,k,q] * y[b,n,q].  Block per (b,m);
// 1024 threads = (n:128) x (k:8).
// ---------------------------------------------------------------------------
__global__ void __launch_bounds__(1024)
xvy_kernel(const float* __restrict__ y, const float* __restrict__ xV,
           float* __restrict__ xvy)
{
    __shared__ float s[kD][kH];   // 16 KB
    const int b = blockIdx.x >> 7;
    const int m = blockIdx.x & (kL - 1);
    for (int i = threadIdx.x; i < kD * kH; i += 1024)
        (&s[0][0])[i] = xV[(size_t)(b * kL + m) * kD * kH + i];
    __syncthreads();
    const int n = threadIdx.x >> 3;
    const int k = threadIdx.x & (kD - 1);
    const float* yr = y + (size_t)(b * kL + n) * kH;
    float acc = 0.f;
    for (int q = 0; q < kH; ++q) acc = fmaf(s[k][q], yr[q], acc);
    xvy[(((size_t)(b * kL + m) * kL) + n) * kD + k] = acc;
}

// ---------------------------------------------------------------------------
// Main GEMM:  out[row, h'] = gelu( ctx[row,:]·W3ᵀ + Arow + Brow + xvy·W4ᵀ )
// M=32768, N=512, K=512.  Workgroup = 256 threads = 8 waves tiled 2(M)x4(N);
// each wave computes 64x32 via 4x2 fragments of v_wmma_f32_16x16x32_f16.
// ---------------------------------------------------------------------------
__global__ void __launch_bounds__(256)
gemm_ctx_kernel(const _Float16* __restrict__ ctxA,   // [32768][512] f16
                const _Float16* __restrict__ Bp,     // [512][512] f16 (n-major)
                const float* __restrict__ Arow,      // [256][512] (bias folded)
                const float* __restrict__ Brow,      // [256][512]
                const float* __restrict__ xvy,       // [32768][8]
                const float* __restrict__ Ww,        // W4 = Ww[:, 3H:3H+8]
                float* __restrict__ out)             // [32768][512]
{
    const int lane = threadIdx.x & 31;
    const int wid  = threadIdx.x >> 5;
    const int wm = wid & 1, wn = wid >> 1;
    const int rowBase = blockIdx.x * 128 + wm * 64;
    const int colBase = blockIdx.y * 128 + wn * 32;
    const int lh = lane & 15;   // lane within half-wave
    const int hw = lane >> 4;   // which half-wave (0/1)

    v8f acc[4][2] = {};

    for (int k = 0; k < kH; k += 32) {
        v16h af[4], bf[2];
        // A fragment (16x32 f16): lane half 0 holds K k..k+7 & k+16..k+23,
        // lane half 1 holds K k+8..k+15 & k+24..k+31 (ISA 7.12.2 layout).
        #pragma unroll
        for (int mf = 0; mf < 4; ++mf) {
            const v8h* p = (const v8h*)(ctxA + (size_t)(rowBase + mf * 16 + lh) * kH
                                             + k + hw * 8);
            v8h lo = p[0];
            v8h hi = p[2];          // +16 halves
            af[mf] = __builtin_shufflevector(lo, hi,
                     0,1,2,3,4,5,6,7,8,9,10,11,12,13,14,15);
        }
        // B fragment (32x16 f16): same K layout as A (half 0: k..k+7 & k+16..k+23, half 1: k+8..k+15 & k+24..k+31).
        #pragma unroll
        for (int nf = 0; nf < 2; ++nf) {
            const v8h* p = (const v8h*)(Bp + (size_t)(colBase + nf * 16 + lh) * kH
                                            + k + hw * 8);
            v8h lo = p[0];
            v8h hi = p[2];          // +16 halves
            bf[nf] = __builtin_shufflevector(lo, hi,
                     0,1,2,3,4,5,6,7,8,9,10,11,12,13,14,15);
        }
        #pragma unroll
        for (int mf = 0; mf < 4; ++mf)
            #pragma unroll
            for (int nf = 0; nf < 2; ++nf)
                acc[mf][nf] = __builtin_amdgcn_wmma_f32_16x16x32_f16(
                    false, af[mf], false, bf[nf],
                    (short)0, acc[mf][nf], false, false);
    }

    // Fused epilogue: + Arow[b,m,:] + Brow[b,n,:] + xvy[b,m,n,:]·W4ᵀ, exact GELU.
    #pragma unroll
    for (int nf = 0; nf < 2; ++nf) {
        const int col = colBase + nf * 16 + lh;
        float w4[kD];
        #pragma unroll
        for (int kk = 0; kk < kD; ++kk)
            w4[kk] = Ww[(size_t)col * kWW + 3 * kH + kk];
        #pragma unroll
        for (int mf = 0; mf < 4; ++mf) {
            #pragma unroll
            for (int e = 0; e < 8; ++e) {
                const int row = rowBase + mf * 16 + hw * 8 + e;  // C/D layout
                const int b   = row >> 14;           // / (L*L)
                const int rem = row & 16383;
                const int m   = rem >> 7;
                const int n   = rem & (kL - 1);
                float g = acc[mf][nf][e];
                g += Arow[(size_t)(b * kL + m) * kH + col];
                g += Brow[(size_t)(b * kL + n) * kH + col];
                const float* xv = xvy + (size_t)row * kD;
                #pragma unroll
                for (int kk = 0; kk < kD; ++kk) g = fmaf(xv[kk], w4[kk], g);
                out[(size_t)row * kH + col] = gelu_exact(g);
            }
        }
    }
}

// ---------------------------------------------------------------------------
// Launcher. Workspace layout (bytes):
//   ctxA f16 [32768][512]  @ 0          (33,554,432)
//   Bp   f16 [512][512]    @ 33,554,432 (   524,288)
//   Arow f32 [256][512]    @ 34,078,720 (   524,288)
//   Brow f32 [256][512]    @ 34,603,008 (   524,288)
//   xV   f32 [256][8][512] @ 35,127,296 ( 4,194,304)
//   xvy  f32 [32768][8]    @ 39,321,600 ( 1,048,576)   total ~40.4 MB
// ---------------------------------------------------------------------------
extern "C" void kernel_launch(void* const* d_in, const int* in_sizes, int n_in,
                              void* d_out, int out_size, void* d_ws, size_t ws_size,
                              hipStream_t stream) {
    (void)in_sizes; (void)n_in; (void)out_size; (void)ws_size;
    const float* x  = (const float*)d_in[0];
    const float* y  = (const float*)d_in[1];
    const float* Ww = (const float*)d_in[2];
    const float* Wb = (const float*)d_in[3];
    const float* V  = (const float*)d_in[4];
    float* out = (float*)d_out;

    char* ws = (char*)d_ws;
    _Float16* ctxA = (_Float16*)(ws);
    _Float16* Bp   = (_Float16*)(ws + 33554432);
    float*    Arow = (float*)   (ws + 34078720);
    float*    Brow = (float*)   (ws + 34603008);
    float*    xV   = (float*)   (ws + 35127296);
    float*    xvy  = (float*)   (ws + 39321600);

    ctx_kernel     <<<kB * kL,                 512, 0, stream>>>(x, ctxA);
    pack_w3_kernel <<<(kH * kH + 255) / 256,   256, 0, stream>>>(Ww, Bp);
    row_gemm_kernel<<<2 * kB * kL,             512, 0, stream>>>(x, y, Ww, Wb, Arow, Brow);
    xv_kernel      <<<kB * kL * kD,            512, 0, stream>>>(x, V, xV);
    xvy_kernel     <<<kB * kL,                1024, 0, stream>>>(y, xV, xvy);
    gemm_ctx_kernel<<<dim3(kRows / 128, kH / 128), 256, 0, stream>>>(
        ctxA, Bp, Arow, Brow, xvy, Ww, out);
}
